// MambaTSAD_41300405518480
// MI455X (gfx1250) — compile-verified, matrix operand order checked by reading, not run on hardware
//
#include <hip/hip_runtime.h>
#include <hip/hip_bf16.h>

// ---------------- model constants ----------------
constexpr int CB   = 4;
constexpr int CL   = 1024;
constexpr int CD   = 512;
constexpr int CNST = 16;
constexpr int CKC  = 4;
constexpr int CDI  = 1024;      // 2*CD
constexpr int CR   = 32;        // (D+15)/16
constexpr int CBL  = CB * CL;   // 4096 rows

typedef _Float16 half_t;
typedef __attribute__((ext_vector_type(16))) _Float16 v16h;
typedef __attribute__((ext_vector_type(8)))  _Float16 v8h;
typedef __attribute__((ext_vector_type(8)))  float    v8f;

// ================= WMMA GEMM =================
// OUT[M,N] = act( A_h[M,K](lda) * W_h[N,K](ldw,+kw_off)^T + bias[N] ) [+ res[M,N]]
// 128x64 tile per 128-thread block (4 waves); each wave owns 32 rows
// (two A fragments) and shares the four B fragments. K stepped by 32.
// Double-buffered LDS; fills are register-staged and branchless.
#define TILE_M 128
#define TILE_N 64
#define TILE_K 32

__device__ inline v16h frag_ld(const half_t* base, int row, int hi) {
  // 16-bit A/B fragment (ISA 7.12.2): lane holds row `row`;
  // halves 0..7 -> K = hi*8 + 0..7, halves 8..15 -> K = 16 + hi*8 + 0..7
  const v8h* p0 = (const v8h*)(base + row * TILE_K + hi * 8);
  const v8h* p1 = (const v8h*)(base + row * TILE_K + 16 + hi * 8);
  v8h lo = *p0;
  v8h hh = *p1;
  return __builtin_shufflevector(lo, hh, 0,1,2,3,4,5,6,7,8,9,10,11,12,13,14,15);
}

template<int ACT, bool RES, bool WF, bool WH>  // ACT: 0=none 1=silu 2=gelu(erf) 3=softplus
__global__ __launch_bounds__(128)
void gemm_wmma_kernel(const half_t* __restrict__ A, int lda,
                      const half_t* __restrict__ W, int ldw, int kw_off,
                      const float* __restrict__ bias,
                      const float* __restrict__ res,
                      float* __restrict__ outF,
                      half_t* __restrict__ outH,
                      int M, int N, int K)
{
  __shared__ __align__(16) half_t lds_a[2][TILE_M * TILE_K];  // 2 x 8 KB
  __shared__ __align__(16) half_t lds_b[2][TILE_N * TILE_K];  // 2 x 4 KB

  const int tid  = threadIdx.x;
  const int wave = tid >> 5;
  const int lane = tid & 31;
  const int lm   = lane & 15;
  const int hi   = lane >> 4;
  const int m0   = blockIdx.y * TILE_M;
  const int n0   = blockIdx.x * TILE_N;

  // fixed (row, 16B-chunk) slot per thread for the cooperative fills
  const int fr = tid >> 2;          // 0..31
  const int fc = (tid & 3) * 8;     // 0,8,16,24 (halves)
  const half_t* aptr = A + (size_t)(m0 + fr) * lda + fc;
  const half_t* bptr = W + (size_t)(n0 + fr) * ldw + kw_off + fc;

  v8h areg[4], breg[2];
  auto g2r = [&](int k0) {
#pragma unroll
    for (int j = 0; j < 4; ++j) areg[j] = *(const v8h*)(aptr + (size_t)(32 * j) * lda + k0);
#pragma unroll
    for (int j = 0; j < 2; ++j) breg[j] = *(const v8h*)(bptr + (size_t)(32 * j) * ldw + k0);
  };
  auto r2s = [&](int buf) {
#pragma unroll
    for (int j = 0; j < 4; ++j) *(v8h*)&lds_a[buf][(32 * j + fr) * TILE_K + fc] = areg[j];
#pragma unroll
    for (int j = 0; j < 2; ++j) *(v8h*)&lds_b[buf][(32 * j + fr) * TILE_K + fc] = breg[j];
  };

  v8f zero = {};
  v8f acc[2][4];
#pragma unroll
  for (int s = 0; s < 2; ++s)
#pragma unroll
    for (int t = 0; t < 4; ++t) acc[s][t] = zero;

  const int nk = K / TILE_K;
  g2r(0);
  r2s(0);
  for (int kt = 0; kt < nk; ++kt) {
    __syncthreads();                       // buf[kt&1] visible to all waves
    if (kt + 1 < nk) g2r((kt + 1) * TILE_K);  // next tile's loads overlap the MMAs
    const half_t* la = lds_a[kt & 1];
    const half_t* lb = lds_b[kt & 1];
    v16h a0 = frag_ld(la, wave * 32 + lm, hi);
    v16h a1 = frag_ld(la, wave * 32 + 16 + lm, hi);
#pragma unroll
    for (int t = 0; t < 4; ++t) {
      v16h bf = frag_ld(lb, t * 16 + lm, hi);
      acc[0][t] = __builtin_amdgcn_wmma_f32_16x16x32_f16(
          false, a0, false, bf, (short)0, acc[0][t], false, false);
      acc[1][t] = __builtin_amdgcn_wmma_f32_16x16x32_f16(
          false, a1, false, bf, (short)0, acc[1][t], false, false);
    }
    // buf[(kt+1)&1] was last read in iteration kt-1; the barrier at the top of
    // this iteration ordered those reads before these writes.
    if (kt + 1 < nk) r2s((kt + 1) & 1);
  }

#pragma unroll
  for (int t = 0; t < 4; ++t) {
    int n = n0 + t * 16 + lm;
    float bv = bias ? bias[n] : 0.0f;
#pragma unroll
    for (int s = 0; s < 2; ++s) {
#pragma unroll
      for (int r = 0; r < 8; ++r) {
        int m = m0 + wave * 32 + s * 16 + r + 8 * hi;
        float v = acc[s][t][r] + bv;
        if (ACT == 1)      v = v / (1.0f + expf(-v));
        else if (ACT == 2) v = 0.5f * v * (1.0f + erff(v * 0.70710678f));
        else if (ACT == 3) v = (v > 20.0f) ? v : log1pf(expf(v));
        size_t o = (size_t)m * N + n;
        if (RES) v += res[o];
        if (WF)  outF[o] = v;
        if (WH)  outH[o] = (half_t)v;
      }
    }
  }
}

template<int ACT, bool RES, bool WF, bool WH>
static inline void gemm_t(hipStream_t s,
                          const half_t* A, int lda,
                          const half_t* W, int ldw, int kwoff,
                          const float* bias, const float* res,
                          float* outF, half_t* outH, int M, int N, int K)
{
  dim3 g(N / TILE_N, M / TILE_M), b(128);
  gemm_wmma_kernel<ACT, RES, WF, WH><<<g, b, 0, s>>>(
      A, lda, W, ldw, kwoff, bias, res, outF, outH, M, N, K);
}

// ================= elementwise / helper kernels =================

__global__ void cvt_f32_f16_kernel(const float* __restrict__ in, half_t* __restrict__ out, int n) {
  int i = blockIdx.x * blockDim.x + threadIdx.x;
  if (i < n) out[i] = (half_t)in[i];
}

// LayerNorm over last dim (D=512); one 128-thread block per row, writes f32 + f16
__global__ __launch_bounds__(128)
void layernorm_kernel(const float* __restrict__ in, const float* __restrict__ w,
                      const float* __restrict__ b, float* __restrict__ outF,
                      half_t* __restrict__ outH)
{
  __shared__ float red[128];
  const int row = blockIdx.x;
  const float* x = in + (size_t)row * CD;
  float s = 0.f;
  for (int j = threadIdx.x; j < CD; j += 128) s += x[j];
  red[threadIdx.x] = s; __syncthreads();
  for (int off = 64; off > 0; off >>= 1) {
    if (threadIdx.x < off) red[threadIdx.x] += red[threadIdx.x + off];
    __syncthreads();
  }
  float mean = red[0] / CD;
  __syncthreads();
  float ss = 0.f;
  for (int j = threadIdx.x; j < CD; j += 128) { float d0 = x[j] - mean; ss += d0 * d0; }
  red[threadIdx.x] = ss; __syncthreads();
  for (int off = 64; off > 0; off >>= 1) {
    if (threadIdx.x < off) red[threadIdx.x] += red[threadIdx.x + off];
    __syncthreads();
  }
  float inv = rsqrtf(red[0] / CD + 1e-5f);
  for (int j = threadIdx.x; j < CD; j += 128) {
    float v = (x[j] - mean) * inv * w[j] + b[j];
    size_t o = (size_t)row * CD + j;
    outF[o] = v;
    outH[o] = (half_t)v;
  }
}

// flip along L:  out[b,l,:] = in[b,L-1-l,:]
__global__ void flip_kernel(const float* __restrict__ in, float* __restrict__ outF,
                            half_t* __restrict__ outH)
{
  int i = blockIdx.x * blockDim.x + threadIdx.x;
  if (i >= CBL * CD) return;
  int d = i % CD, bl = i / CD, b = bl / CL, l = bl % CL;
  float v = in[((size_t)(b * CL + (CL - 1 - l))) * CD + d];
  outF[i] = v;
  outH[i] = (half_t)v;
}

// causal depthwise conv (KC=4) over xm = xz[:, :DI] + bias, then SiLU
__global__ void conv_causal_kernel(const float* __restrict__ xz, const float* __restrict__ w,
                                   const float* __restrict__ bias,
                                   float* __restrict__ outF, half_t* __restrict__ outH)
{
  int i = blockIdx.x * blockDim.x + threadIdx.x;
  if (i >= CBL * CDI) return;
  int c = i % CDI, bl = i / CDI, b = bl / CL, l = bl % CL;
  float s = bias[c];
#pragma unroll
  for (int j = 0; j < CKC; ++j) {
    int ll = l - (CKC - 1) + j;
    if (ll >= 0) s += xz[((size_t)(b * CL + ll)) * (2 * CDI) + c] * w[c * CKC + j];
  }
  float v = s / (1.0f + expf(-s));
  outF[i] = v;
  outH[i] = (half_t)v;
}

// selective scan: one thread per (b, channel); h[16] recurrence over L
__global__ void scan_kernel(const float* __restrict__ dt, const float* __restrict__ xact,
                            const float* __restrict__ xdb, const float* __restrict__ A_log,
                            float* __restrict__ y)
{
  int t = blockIdx.x * blockDim.x + threadIdx.x;
  if (t >= CB * CDI) return;
  int b = t / CDI, c = t % CDI;
  float Aa[CNST], h[CNST];
#pragma unroll
  for (int n = 0; n < CNST; ++n) { Aa[n] = -expf(A_log[c * CNST + n]); h[n] = 0.f; }
  for (int l = 0; l < CL; ++l) {
    size_t idx = (size_t)(b * CL + l);
    float dtv = dt[idx * CDI + c];
    float xv  = xact[idx * CDI + c];
    const float* row = xdb + idx * (CR + 2 * CNST);
    if (l + 1 < CL) {
      __builtin_prefetch(dt + (idx + 1) * CDI + c, 0, 0);
      __builtin_prefetch(xact + (idx + 1) * CDI + c, 0, 0);
    }
    float dx = dtv * xv;
    float accv = 0.f;
#pragma unroll
    for (int n = 0; n < CNST; ++n) {
      h[n] = h[n] * expf(dtv * Aa[n]) + dx * row[CR + n];
      accv += h[n] * row[CR + CNST + n];
    }
    y[idx * CDI + c] = accv;
  }
}

// gate: y = (yscan + xact * Dvec) * silu(z),  z = xz[:, DI:2DI]
__global__ void gate_kernel(const float* __restrict__ yscan, const float* __restrict__ xact,
                            const float* __restrict__ Dvec, const float* __restrict__ xz,
                            float* __restrict__ outF, half_t* __restrict__ outH)
{
  int i = blockIdx.x * blockDim.x + threadIdx.x;
  if (i >= CBL * CDI) return;
  int c = i % CDI; size_t row = (size_t)(i / CDI);
  float z = xz[row * (2 * CDI) + CDI + c];
  float sz = z / (1.0f + expf(-z));
  float v = (yscan[i] + xact[i] * Dvec[c]) * sz;
  outF[i] = v;
  outH[i] = (half_t)v;
}

// g += 0.5*(y_fwd + flip_L(y_bwd))
__global__ void comb_kernel(float* __restrict__ g, const float* __restrict__ yf,
                            const float* __restrict__ yb)
{
  int i = blockIdx.x * blockDim.x + threadIdx.x;
  if (i >= CBL * CD) return;
  int d = i % CD, bl = i / CD, b = bl / CL, l = bl % CL;
  g[i] += 0.5f * (yf[i] + yb[((size_t)(b * CL + (CL - 1 - l))) * CD + d]);
}

// instance norm over L per (b,d), optional SiLU; one block per batch, thread=channel
__global__ __launch_bounds__(512)
void instnorm_l_kernel(const float* __restrict__ in, float* __restrict__ outF,
                       half_t* __restrict__ outH, int do_silu)
{
  int b = blockIdx.x, d = threadIdx.x;
  float s = 0.f, ss = 0.f;
  for (int l = 0; l < CL; ++l) {
    float v = in[((size_t)(b * CL + l)) * CD + d];
    s += v; ss += v * v;
  }
  float m = s / CL;
  float inv = rsqrtf(ss / CL - m * m + 1e-5f);
  for (int l = 0; l < CL; ++l) {
    size_t idx = ((size_t)(b * CL + l)) * CD + d;
    float v = (in[idx] - m) * inv;
    if (do_silu) v = v / (1.0f + expf(-v));
    if (outF) outF[idx] = v;
    if (outH) outH[idx] = (half_t)v;
  }
}

// same-pad depthwise conv over L, kernel width k (5 or 7)
__global__ void conv_same_kernel(const float* __restrict__ in, const float* __restrict__ w,
                                 float* __restrict__ out, int k)
{
  int i = blockIdx.x * blockDim.x + threadIdx.x;
  if (i >= CBL * CD) return;
  int c = i % CD, bl = i / CD, b = bl / CL, l = bl % CL;
  float s = 0.f;
  for (int j = 0; j < k; ++j) {
    int ll = l - k / 2 + j;
    if (ll >= 0 && ll < CL) s += in[((size_t)(b * CL + ll)) * CD + c] * w[c * k + j];
  }
  out[i] = s;
}

// ================= host orchestration =================

static inline int ceil_div(int a, int b) { return (a + b - 1) / b; }

extern "C" void kernel_launch(void* const* d_in, const int* in_sizes, int n_in,
                              void* d_out, int out_size, void* d_ws, size_t ws_size,
                              hipStream_t stream)
{
  (void)in_sizes; (void)n_in; (void)out_size; (void)ws_size;

  // -------- inputs (setup_inputs order) --------
  const float* x_f       = (const float*)d_in[0];
  const float* in_w      = (const float*)d_in[1];   // (4,2048,512)
  const float* in_b      = (const float*)d_in[2];   // (4,2048)
  const float* conv_w    = (const float*)d_in[3];   // (4,1024,4)
  const float* conv_b    = (const float*)d_in[4];   // (4,1024)
  const float* xp_w      = (const float*)d_in[5];   // (4,64,1024)
  const float* dt_w      = (const float*)d_in[6];   // (4,1024,32)
  const float* dt_b      = (const float*)d_in[7];   // (4,1024)
  const float* A_log     = (const float*)d_in[8];   // (4,1024,16)
  const float* D_par     = (const float*)d_in[9];   // (4,1024)
  const float* out_w     = (const float*)d_in[10];  // (4,512,1024)
  const float* out_b     = (const float*)d_in[11];  // (4,512)
  const float* ln_w      = (const float*)d_in[12];  // (2,512)
  const float* ln_b      = (const float*)d_in[13];
  const float* ffn_ln_w  = (const float*)d_in[14];
  const float* ffn_ln_b  = (const float*)d_in[15];
  const float* ffn_w1    = (const float*)d_in[16];  // (2,2048,512)
  const float* ffn_b1    = (const float*)d_in[17];
  const float* ffn_w2    = (const float*)d_in[18];  // (2,512,2048)
  const float* ffn_b2    = (const float*)d_in[19];
  const float* l5_pw1_w  = (const float*)d_in[20];
  const float* l5_pw1_b  = (const float*)d_in[21];
  const float* l5_dw_w   = (const float*)d_in[22];
  const float* l5_pw2_w  = (const float*)d_in[23];
  const float* l5_pw2_b  = (const float*)d_in[24];
  const float* l7_pw1_w  = (const float*)d_in[25];
  const float* l7_pw1_b  = (const float*)d_in[26];
  const float* l7_dw_w   = (const float*)d_in[27];
  const float* l7_pw2_w  = (const float*)d_in[28];
  const float* l7_pw2_b  = (const float*)d_in[29];
  const float* proj_w    = (const float*)d_in[30];  // (512,1536)
  const float* proj_b    = (const float*)d_in[31];
  float* out_f = (float*)d_out;

  // -------- workspace carve-up --------
  size_t off = 0;
  auto alloc = [&](size_t bytes) -> void* {
    void* p = (char*)d_ws + off;
    off += (bytes + 255) & ~(size_t)255;
    return p;
  };
  const size_t RD  = (size_t)CBL * CD;       // 4096*512
  const size_t RDI = (size_t)CBL * CDI;      // 4096*1024
  const size_t R2  = (size_t)CBL * 2 * CDI;  // 4096*2048

  float*  g      = (float*) alloc(RD * 4);
  float*  xn     = (float*) alloc(RD * 4);
  half_t* xn_h   = (half_t*)alloc(RD * 2);
  float*  xnf    = (float*) alloc(RD * 4);
  half_t* xnf_h  = (half_t*)alloc(RD * 2);
  float*  yf     = (float*) alloc(RD * 4);
  float*  yb     = (float*) alloc(RD * 4);
  float*  bufA   = (float*) alloc(R2 * 4);   // xz / ffn hidden
  half_t* bufAh  = (half_t*)alloc(R2 * 2);
  float*  bufB   = (float*) alloc(RDI * 4);  // xm_act / local tmp
  half_t* bufBh  = (half_t*)alloc(RDI * 2);
  float*  bufC   = (float*) alloc(RDI * 4);  // dt / local tmp / t1
  float*  bufD   = (float*) alloc(RDI * 4);  // y_scan,y / t2
  half_t* bufDh  = (half_t*)alloc(RDI * 2);
  float*  xdb    = (float*) alloc((size_t)CBL * 64 * 4);
  half_t* xdb_h  = (half_t*)alloc((size_t)CBL * 64 * 2);
  float*  l5o    = (float*) alloc(RD * 4);
  half_t* l5h    = (half_t*)alloc(RD * 2);
  float*  l7o    = (float*) alloc(RD * 4);
  half_t* l7h    = (half_t*)alloc(RD * 2);
  half_t* x_h    = (half_t*)alloc(RD * 2);
  half_t* g_h    = (half_t*)alloc(RD * 2);
  // f16 weight copies
  half_t* in_w_h  = (half_t*)alloc((size_t)4 * 2048 * 512 * 2);
  half_t* xp_w_h  = (half_t*)alloc((size_t)4 * 64 * 1024 * 2);
  half_t* dt_w_h  = (half_t*)alloc((size_t)4 * 1024 * 32 * 2);
  half_t* out_w_h = (half_t*)alloc((size_t)4 * 512 * 1024 * 2);
  half_t* w1_h    = (half_t*)alloc((size_t)2 * 2048 * 512 * 2);
  half_t* w2_h    = (half_t*)alloc((size_t)2 * 512 * 2048 * 2);
  half_t* p15_h   = (half_t*)alloc((size_t)512 * 512 * 2);
  half_t* p25_h   = (half_t*)alloc((size_t)512 * 512 * 2);
  half_t* p17_h   = (half_t*)alloc((size_t)512 * 512 * 2);
  half_t* p27_h   = (half_t*)alloc((size_t)512 * 512 * 2);
  half_t* proj_h  = (half_t*)alloc((size_t)512 * 1536 * 2);

  auto cvt = [&](const float* src, half_t* dst, int n) {
    cvt_f32_f16_kernel<<<ceil_div(n, 256), 256, 0, stream>>>(src, dst, n);
  };

  // -------- weight + input conversions (f32 -> f16) --------
  cvt(in_w,     in_w_h,  4 * 2048 * 512);
  cvt(xp_w,     xp_w_h,  4 * 64 * 1024);
  cvt(dt_w,     dt_w_h,  4 * 1024 * 32);
  cvt(out_w,    out_w_h, 4 * 512 * 1024);
  cvt(ffn_w1,   w1_h,    2 * 2048 * 512);
  cvt(ffn_w2,   w2_h,    2 * 512 * 2048);
  cvt(l5_pw1_w, p15_h,   512 * 512);
  cvt(l5_pw2_w, p25_h,   512 * 512);
  cvt(l7_pw1_w, p17_h,   512 * 512);
  cvt(l7_pw2_w, p27_h,   512 * 512);
  cvt(proj_w,   proj_h,  512 * 1536);
  cvt(x_f,      x_h,     (int)RD);

  // g = x
  hipMemcpyAsync(g, x_f, RD * 4, hipMemcpyDeviceToDevice, stream);

  const int EW_D  = ceil_div((int)RD, 256);
  const int EW_DI = ceil_div((int)RDI, 256);

  // ==================== two mamba/FFN layers ====================
  for (int i = 0; i < 2; ++i) {
    layernorm_kernel<<<CBL, 128, 0, stream>>>(g, ln_w + i * CD, ln_b + i * CD, xn, xn_h);
    flip_kernel<<<EW_D, 256, 0, stream>>>(xn, xnf, xnf_h);

    for (int dir = 0; dir < 2; ++dir) {
      const int m = 2 * i + dir;
      const half_t* Ain  = dir ? xnf_h : xn_h;
      float*        yout = dir ? yb : yf;

      // xz = Ain @ in_w[m]^T + in_b[m]           [BL, 2048]
      gemm_t<0, false, true, false>(stream, Ain, CD,
          in_w_h + (size_t)m * 2048 * 512, CD, 0,
          in_b + (size_t)m * 2048, nullptr, bufA, nullptr, CBL, 2 * CDI, CD);
      // xm = silu(causal_conv(xz[:, :DI]) + conv_b)
      conv_causal_kernel<<<EW_DI, 256, 0, stream>>>(
          bufA, conv_w + (size_t)m * CDI * CKC, conv_b + (size_t)m * CDI, bufB, bufBh);
      // xdb = xm @ xp_w[m]^T                     [BL, 64]
      gemm_t<0, false, true, true>(stream, bufBh, CDI,
          xp_w_h + (size_t)m * 64 * 1024, CDI, 0,
          nullptr, nullptr, xdb, xdb_h, CBL, 64, CDI);
      // dt = softplus(xdb[:, :32] @ dt_w[m]^T + dt_b[m])   [BL, 1024]
      gemm_t<3, false, true, false>(stream, xdb_h, 64,
          dt_w_h + (size_t)m * 1024 * 32, 32, 0,
          dt_b + (size_t)m * CDI, nullptr, bufC, nullptr, CBL, CDI, CR);
      // selective scan
      scan_kernel<<<ceil_div(CB * CDI, 256), 256, 0, stream>>>(
          bufC, bufB, xdb, A_log + (size_t)m * CDI * CNST, bufD);
      // y = (y_scan + xm*D) * silu(z)
      gate_kernel<<<EW_DI, 256, 0, stream>>>(
          bufD, bufB, D_par + (size_t)m * CDI, bufA, bufD, bufDh);
      // y_out = y @ out_w[m]^T + out_b[m]        [BL, 512]
      gemm_t<0, false, true, false>(stream, bufDh, CDI,
          out_w_h + (size_t)m * 512 * 1024, CDI, 0,
          out_b + (size_t)m * CD, nullptr, yout, nullptr, CBL, CD, CDI);
    }
    // g += 0.5*(y_fwd + flip(y_bwd))
    comb_kernel<<<EW_D, 256, 0, stream>>>(g, yf, yb);

    // FFN: hn = LN(g); h = gelu(hn@w1^T+b1)@w2^T + b2; g += h
    layernorm_kernel<<<CBL, 128, 0, stream>>>(g, ffn_ln_w + i * CD, ffn_ln_b + i * CD, xn, xn_h);
    gemm_t<2, false, false, true>(stream, xn_h, CD,
        w1_h + (size_t)i * 2048 * 512, CD, 0,
        ffn_b1 + (size_t)i * 2048, nullptr, nullptr, bufAh, CBL, 4 * CD, CD);
    gemm_t<0, true, true, false>(stream, bufAh, 4 * CD,
        w2_h + (size_t)i * 512 * 2048, 4 * CD, 0,
        ffn_b2 + (size_t)i * CD, g, g, nullptr, CBL, CD, 4 * CD);
  }

  // ==================== local conv branches (on x) ====================
  struct LocP { const half_t* pw1; const float* pw1b; const float* dw;
                const half_t* pw2; const float* pw2b; int k; float* out; half_t* outh; };
  LocP locs[2] = {
    { p15_h, l5_pw1_b, l5_dw_w, p25_h, l5_pw2_b, 5, l5o, l5h },
    { p17_h, l7_pw1_b, l7_dw_w, p27_h, l7_pw2_b, 7, l7o, l7h },
  };
  for (int j = 0; j < 2; ++j) {
    const LocP& P = locs[j];
    // h = x @ pw1^T + b
    gemm_t<0, false, true, false>(stream, x_h, CD, P.pw1, CD, 0,
        P.pw1b, nullptr, bufC, nullptr, CBL, CD, CD);
    // silu(instnorm over L)
    instnorm_l_kernel<<<CB, CD, 0, stream>>>(bufC, bufB, nullptr, 1);
    // depthwise same-pad conv
    conv_same_kernel<<<EW_D, 256, 0, stream>>>(bufB, P.dw, bufC, P.k);
    // silu(instnorm over L) -> f16
    instnorm_l_kernel<<<CB, CD, 0, stream>>>(bufC, nullptr, bufBh, 1);
    // out = h @ pw2^T + b
    gemm_t<0, false, true, true>(stream, bufBh, CD, P.pw2, CD, 0,
        P.pw2b, nullptr, P.out, P.outh, CBL, CD, CD);
  }

  // ==================== final projection + residual ====================
  cvt(g, g_h, (int)RD);
  // out = g@Wp[:, :512]^T + proj_b + x
  gemm_t<0, true, true, false>(stream, g_h, CD, proj_h, 3 * CD, 0,
      proj_b, x_f, bufC, nullptr, CBL, CD, CD);
  // out += l5@Wp[:,512:1024]^T
  gemm_t<0, true, true, false>(stream, l5h, CD, proj_h, 3 * CD, CD,
      nullptr, bufC, bufD, nullptr, CBL, CD, CD);
  // out += l7@Wp[:,1024:1536]^T  -> d_out
  gemm_t<0, true, true, false>(stream, l7h, CD, proj_h, 3 * CD, 2 * CD,
      nullptr, bufD, out_f, nullptr, CBL, CD, CD);
}
